// SAGEWithCats_22247930593832
// MI455X (gfx1250) — matrix operand order repaired and checked by reading, not compiled, not run on hardware
//
#include <hip/hip_runtime.h>
#include <hip/hip_bf16.h>

typedef __attribute__((ext_vector_type(16))) _Float16 v16h;
typedef __attribute__((ext_vector_type(8)))  _Float16 v8h;
typedef __attribute__((ext_vector_type(8)))  float    v8f;

#define HID 128
#define EPS 1e-5f

// ---------------------------------------------------------------------------
// Transpose 128x128 f32 weight -> f16 Wt[n][k] = W[k][n]
// ---------------------------------------------------------------------------
__global__ void k_transpose_f16(const float* __restrict__ w, _Float16* __restrict__ wt) {
    int t = blockIdx.x * blockDim.x + threadIdx.x;   // 0..16383
    int k = t >> 7, n = t & 127;
    wt[n * HID + k] = (_Float16)w[t];
}

// ---------------------------------------------------------------------------
// Fused embedding gather + input linear (71 -> 128) + ReLU
// block = 128 threads, one node per block
// ---------------------------------------------------------------------------
__global__ void k_input(const float* __restrict__ x_num, const int* __restrict__ x_cat,
                        const float* __restrict__ e0, const float* __restrict__ e1,
                        const float* __restrict__ e2, const float* __restrict__ e3,
                        const float* __restrict__ w_in, const float* __restrict__ b_in,
                        float* __restrict__ x, _Float16* __restrict__ xH, int n) {
    int i = blockIdx.x;
    if (i >= n) return;
    int t = threadIdx.x;
    __shared__ float in[72];
    if (t < 32) {
        in[t] = x_num[(size_t)i * 32 + t];
    } else if (t < 71) {
        int c0 = x_cat[i * 4 + 0], c1 = x_cat[i * 4 + 1];
        int c2 = x_cat[i * 4 + 2], c3 = x_cat[i * 4 + 3];
        float v;
        if (t < 42)      v = e0[c0 * 10 + (t - 32)];
        else if (t < 48) v = e1[c1 * 6  + (t - 42)];
        else if (t < 53) v = e2[c2 * 5  + (t - 48)];
        else             v = e3[c3 * 18 + (t - 53)];
        in[t] = v;
    }
    __syncthreads();
    float acc = b_in[t];
    #pragma unroll 1
    for (int k = 0; k < 71; ++k)
        acc = fmaf(in[k], w_in[k * HID + t], acc);
    acc = fmaxf(acc, 0.0f);
    x [(size_t)i * HID + t] = acc;
    xH[(size_t)i * HID + t] = (_Float16)acc;
}

// ---------------------------------------------------------------------------
// In-degree count (float, exact for small counts)
// ---------------------------------------------------------------------------
__global__ void k_degree(const int* __restrict__ dst, float* __restrict__ cnt, int E) {
    int e = blockIdx.x * blockDim.x + threadIdx.x;
    if (e < E) atomicAdd(&cnt[dst[e]], 1.0f);
}

// ---------------------------------------------------------------------------
// Edge scatter: one wave per edge; lane handles 4 contiguous features.
// Coalesced 512B row gather + float atomic scatter-add.
// ---------------------------------------------------------------------------
__global__ void k_scatter(const float* __restrict__ x, const int* __restrict__ src,
                          const int* __restrict__ dst, float* __restrict__ agg, int E) {
    int w = (blockIdx.x * blockDim.x + threadIdx.x) >> 5;
    int lane = threadIdx.x & 31;
    if (w >= E) return;
    int s = src[w], d = dst[w];
    const float4 v = *(const float4*)(x + (size_t)s * HID + lane * 4);
    float* ap = agg + (size_t)d * HID + lane * 4;
    atomicAdd(ap + 0, v.x);
    atomicAdd(ap + 1, v.y);
    atomicAdd(ap + 2, v.z);
    atomicAdd(ap + 3, v.w);
}

// ---------------------------------------------------------------------------
// mean = agg / max(cnt,1)  -> f16
// ---------------------------------------------------------------------------
__global__ void k_mean(const float* __restrict__ agg, const float* __restrict__ cnt,
                       _Float16* __restrict__ meanH, int n) {
    int t = blockIdx.x * blockDim.x + threadIdx.x;
    if (t >= n * HID) return;
    float c = cnt[t >> 7];
    c = c < 1.0f ? 1.0f : c;
    meanH[t] = (_Float16)(agg[t] / c);
}

// ---------------------------------------------------------------------------
// WMMA SAGE conv:  raw = meanH @ Wl + xH @ Wr + bl
//
// block = 256 threads (8 waves); block tile = 32 nodes x 128 cols.
// A tiles (mean, x: 32x128 f16 each) are staged in LDS once per block via
// CDNA5 async global->LDS copies (ASYNCcnt + s_wait_asynccnt), removing the
// 8x per-wave redundant global A traffic. Each wave owns a 16-col strip and
// two 16x16 output tiles; B fragments load once per K-step and are reused
// across both row tiles -> 16 v_wmma_f32_16x16x32_f16 per wave.
//
// Fragment layouts per CDNA5 ISA 7.12.2 (wave32):
//   A 16x32 f16: lane<16 -> row=lane, K {0..7, 16..23}; lane>=16 -> K {8..15, 24..31}
//   B 32x16 f16: lane<16 -> col=lane, K {0..7, 16..23}; lane>=16 -> K {8..15, 24..31}
//   C/D 16x16 f32: lane(hi,lr): VGPR r -> (M = hi*8+r, N = lr)
// Wt is stored transposed (Wt[n][k]) so B loads are two contiguous v8h chunks per lane.
// ---------------------------------------------------------------------------
__device__ __forceinline__ v16h load_Afrag(const _Float16* row, int k0, int hi) {
    v8h alo = *(const v8h*)(row + k0 + hi * 8);
    v8h ahi = *(const v8h*)(row + k0 + 16 + hi * 8);
    v16h a;
    #pragma unroll
    for (int i = 0; i < 8; ++i) { a[i] = alo[i]; a[i + 8] = ahi[i]; }
    return a;
}

__global__ void k_conv_wmma(const _Float16* __restrict__ xH,
                            const _Float16* __restrict__ meanH,
                            const _Float16* __restrict__ wltH,   // Wl^T [128][128]
                            const _Float16* __restrict__ wrtH,   // Wr^T [128][128]
                            const float* __restrict__ bl,
                            float* __restrict__ raw, int n) {
    __shared__ _Float16 sM[32 * HID];   // mean tile, 8 KB
    __shared__ _Float16 sX[32 * HID];   // x    tile, 8 KB

    const int t    = threadIdx.x;
    const int wave = t >> 5;            // 0..7 -> 16-col strip
    const int lane = t & 31;
    const int lr = lane & 15, hi = lane >> 4;
    const int row0 = blockIdx.x * 32;
    const int col0 = wave * 16;

    // ---- async copy 32x128 f16 tiles to LDS (32B per thread per matrix) ----
    {
        const int r = t >> 3;           // 0..31  tile row
        const int c = (t & 7) * 16;     // halves: 8 chunks x 16 halves = 128
        const size_t goff = (size_t)(row0 + r) * HID + c;
        unsigned lds_m = (unsigned)(size_t)(&sM[r * HID + c]);
        unsigned lds_x = (unsigned)(size_t)(&sX[r * HID + c]);
        unsigned long long gm = (unsigned long long)(const void*)(meanH + goff);
        unsigned long long gx = (unsigned long long)(const void*)(xH + goff);
        asm volatile("global_load_async_to_lds_b128 %0, %1, off"
                     :: "v"(lds_m), "v"(gm) : "memory");
        asm volatile("global_load_async_to_lds_b128 %0, %1, off offset:16"
                     :: "v"(lds_m), "v"(gm) : "memory");
        asm volatile("global_load_async_to_lds_b128 %0, %1, off"
                     :: "v"(lds_x), "v"(gx) : "memory");
        asm volatile("global_load_async_to_lds_b128 %0, %1, off offset:16"
                     :: "v"(lds_x), "v"(gx) : "memory");
        asm volatile("s_wait_asynccnt 0x0" ::: "memory");
    }
    __syncthreads();

    const _Float16* mrow0 = &sM[lr * HID];
    const _Float16* mrow1 = &sM[(16 + lr) * HID];
    const _Float16* xrow0 = &sX[lr * HID];
    const _Float16* xrow1 = &sX[(16 + lr) * HID];
    const _Float16* bcol_l = wltH + (size_t)(col0 + lr) * HID;
    const _Float16* bcol_r = wrtH + (size_t)(col0 + lr) * HID;

    v8f acc0 = {}, acc1 = {};
    #pragma unroll
    for (int k0 = 0; k0 < HID; k0 += 32) {
        v16h bm = load_Afrag(bcol_l, k0, hi);
        v16h bx = load_Afrag(bcol_r, k0, hi);

        v16h am0 = load_Afrag(mrow0, k0, hi);
        acc0 = __builtin_amdgcn_wmma_f32_16x16x32_f16(false, am0, false, bm,
                                                      (short)0, acc0, false, false);
        v16h ax0 = load_Afrag(xrow0, k0, hi);
        acc0 = __builtin_amdgcn_wmma_f32_16x16x32_f16(false, ax0, false, bx,
                                                      (short)0, acc0, false, false);
        v16h am1 = load_Afrag(mrow1, k0, hi);
        acc1 = __builtin_amdgcn_wmma_f32_16x16x32_f16(false, am1, false, bm,
                                                      (short)0, acc1, false, false);
        v16h ax1 = load_Afrag(xrow1, k0, hi);
        acc1 = __builtin_amdgcn_wmma_f32_16x16x32_f16(false, ax1, false, bx,
                                                      (short)0, acc1, false, false);
    }

    float bias = bl[col0 + lr];
    #pragma unroll
    for (int r = 0; r < 8; ++r) {
        int row = row0 + hi * 8 + r;
        if (row < n) raw[(size_t)row * HID + col0 + lr] = acc0[r] + bias;
        int row2 = row + 16;
        if (row2 < n) raw[(size_t)row2 * HID + col0 + lr] = acc1[r] + bias;
    }
}

// ---------------------------------------------------------------------------
// LayerNorm + ReLU + residual:  x += 0.5 * relu(LN(raw));  refresh xH
// block = 128 threads, one node per block
// ---------------------------------------------------------------------------
__global__ void k_ln_res(const float* __restrict__ raw, float* __restrict__ x,
                         _Float16* __restrict__ xH,
                         const float* __restrict__ g, const float* __restrict__ be, int n) {
    int i = blockIdx.x;
    if (i >= n) return;
    int t = threadIdx.x;
    __shared__ float sm[HID];
    float v = raw[(size_t)i * HID + t];
    sm[t] = v;
    __syncthreads();
    #pragma unroll
    for (int s = 64; s > 0; s >>= 1) { if (t < s) sm[t] += sm[t + s]; __syncthreads(); }
    float mu = sm[0] * (1.0f / HID);
    __syncthreads();
    float d = v - mu;
    sm[t] = d * d;
    __syncthreads();
    #pragma unroll
    for (int s = 64; s > 0; s >>= 1) { if (t < s) sm[t] += sm[t + s]; __syncthreads(); }
    float var = sm[0] * (1.0f / HID);
    float y = d * __frsqrt_rn(var + EPS) * g[t] + be[t];
    y = fmaxf(y, 0.0f);
    float xn = x[(size_t)i * HID + t] + 0.5f * y;
    x [(size_t)i * HID + t] = xn;
    xH[(size_t)i * HID + t] = (_Float16)xn;
}

// ---------------------------------------------------------------------------
// Head: h = relu(x @ wh1 + bh1) [128->64]; out = h @ wh2 + bh2 [64->1]
// block = 64 threads, one node per block
// ---------------------------------------------------------------------------
__global__ void k_head(const float* __restrict__ x, const float* __restrict__ wh1,
                       const float* __restrict__ bh1, const float* __restrict__ wh2,
                       const float* __restrict__ bh2, float* __restrict__ out, int n) {
    int i = blockIdx.x;
    if (i >= n) return;
    int j = threadIdx.x;
    const float* xi = x + (size_t)i * HID;
    float acc = bh1[j];
    #pragma unroll 4
    for (int k = 0; k < HID; ++k)
        acc = fmaf(xi[k], wh1[k * 64 + j], acc);
    float h = fmaxf(acc, 0.0f);
    __shared__ float sm[64];
    sm[j] = h * wh2[j];
    __syncthreads();
    #pragma unroll
    for (int s = 32; s > 0; s >>= 1) { if (j < s) sm[j] += sm[j + s]; __syncthreads(); }
    if (j == 0) out[i] = sm[0] + bh2[0];
}

// ---------------------------------------------------------------------------
static inline size_t align256(size_t v) { return (v + 255) & ~(size_t)255; }

extern "C" void kernel_launch(void* const* d_in, const int* in_sizes, int n_in,
                              void* d_out, int out_size, void* d_ws, size_t ws_size,
                              hipStream_t stream) {
    const float* x_num = (const float*)d_in[0];
    const int*   x_cat = (const int*)d_in[1];
    const int*   eidx  = (const int*)d_in[2];
    const float* e0    = (const float*)d_in[3];
    const float* e1    = (const float*)d_in[4];
    const float* e2    = (const float*)d_in[5];
    const float* e3    = (const float*)d_in[6];
    const float* w_in  = (const float*)d_in[7];
    const float* b_in  = (const float*)d_in[8];
    const float* w1l   = (const float*)d_in[9];
    const float* b1l   = (const float*)d_in[10];
    const float* w1r   = (const float*)d_in[11];
    const float* w2l   = (const float*)d_in[12];
    const float* b2l   = (const float*)d_in[13];
    const float* w2r   = (const float*)d_in[14];
    const float* g1    = (const float*)d_in[15];
    const float* be1   = (const float*)d_in[16];
    const float* g2    = (const float*)d_in[17];
    const float* be2   = (const float*)d_in[18];
    const float* wh1   = (const float*)d_in[19];
    const float* bh1   = (const float*)d_in[20];
    const float* wh2   = (const float*)d_in[21];
    const float* bh2   = (const float*)d_in[22];

    const int N = in_sizes[0] / 32;
    const int E = in_sizes[2] / 2;
    const int* src = eidx;
    const int* dst = eidx + E;

    // workspace layout
    char* p = (char*)d_ws;
    float*    x    = (float*)p;    p += align256((size_t)N * HID * 4);
    float*    agg  = (float*)p;    p += align256((size_t)N * HID * 4);   // reused as `raw`
    _Float16* xH   = (_Float16*)p; p += align256((size_t)N * HID * 2);
    _Float16* mH   = (_Float16*)p; p += align256((size_t)N * HID * 2);
    float*    cnt  = (float*)p;    p += align256((size_t)N * 4);
    _Float16* wlt1 = (_Float16*)p; p += align256((size_t)HID * HID * 2);
    _Float16* wrt1 = (_Float16*)p; p += align256((size_t)HID * HID * 2);
    _Float16* wlt2 = (_Float16*)p; p += align256((size_t)HID * HID * 2);
    _Float16* wrt2 = (_Float16*)p; p += align256((size_t)HID * HID * 2);

    // one-time weight transposes to f16
    k_transpose_f16<<<64, 256, 0, stream>>>(w1l, wlt1);
    k_transpose_f16<<<64, 256, 0, stream>>>(w1r, wrt1);
    k_transpose_f16<<<64, 256, 0, stream>>>(w2l, wlt2);
    k_transpose_f16<<<64, 256, 0, stream>>>(w2r, wrt2);

    // input MLP
    k_input<<<N, HID, 0, stream>>>(x_num, x_cat, e0, e1, e2, e3, w_in, b_in, x, xH, N);

    // degrees (once)
    hipMemsetAsync(cnt, 0, (size_t)N * 4, stream);
    k_degree<<<(E + 255) / 256, 256, 0, stream>>>(dst, cnt, E);

    const int convBlocks    = (N + 31) / 32;
    const int scatterBlocks = (E + 7) / 8;     // one wave per edge
    const int meanBlocks    = (int)(((size_t)N * HID + 255) / 256);

    // ---- conv layer 1 ----
    hipMemsetAsync(agg, 0, (size_t)N * HID * 4, stream);
    k_scatter<<<scatterBlocks, 256, 0, stream>>>(x, src, dst, agg, E);
    k_mean<<<meanBlocks, 256, 0, stream>>>(agg, cnt, mH, N);
    k_conv_wmma<<<convBlocks, 256, 0, stream>>>(xH, mH, wlt1, wrt1, b1l, agg, N);
    k_ln_res<<<N, HID, 0, stream>>>(agg, x, xH, g1, be1, N);

    // ---- conv layer 2 ----
    hipMemsetAsync(agg, 0, (size_t)N * HID * 4, stream);
    k_scatter<<<scatterBlocks, 256, 0, stream>>>(x, src, dst, agg, E);
    k_mean<<<meanBlocks, 256, 0, stream>>>(agg, cnt, mH, N);
    k_conv_wmma<<<convBlocks, 256, 0, stream>>>(xH, mH, wlt2, wrt2, b2l, agg, N);
    k_ln_res<<<N, HID, 0, stream>>>(agg, x, xH, g2, be2, N);

    // head
    k_head<<<N, 64, 0, stream>>>(x, wh1, bh1, wh2, bh2, (float*)d_out, N);
}
